// APPNPStack_52639119180062
// MI455X (gfx1250) — compile-verified, matrix operand order checked by reading, not run on hardware
//
#include <hip/hip_runtime.h>

typedef __attribute__((ext_vector_type(16))) __bf16 v16bf;
typedef __attribute__((ext_vector_type(8)))  __bf16 v8bf;
typedef __attribute__((ext_vector_type(4)))  __bf16 v4bf;
typedef __attribute__((ext_vector_type(8)))  float  v8f;

#define N_NODES 100000   // divisible by 16 -> no M-tail anywhere
#define D_IN    256
#define HID     256
#define D_OUT   64
#define K_HOPS  10
#define ALPHA   0.1f
#define BN_EPS  1e-5f

__device__ __forceinline__ __bf16 f2bf(float f) {
  union { float f; unsigned u; } v; v.f = f;
  unsigned r = v.u + 0x7FFFu + ((v.u >> 16) & 1u);   // round-to-nearest-even
  union { unsigned short s; __bf16 b; } o; o.s = (unsigned short)(r >> 16);
  return o.b;
}
__device__ __forceinline__ float bf2f(__bf16 b) {
  union { __bf16 b; unsigned short s; } i; i.b = b;
  union { unsigned u; float f; } o; o.u = ((unsigned)i.s) << 16;
  return o.f;
}

#define WMMA_BF16(A, B, C) \
  __builtin_amdgcn_wmma_f32_16x16x32_bf16(false, (A), false, (B), (short)0, (C), false, false)

#define CAT16(lo, hi) \
  __builtin_shufflevector((lo), (hi), 0,1,2,3,4,5,6,7,8,9,10,11,12,13,14,15)

// -------- f32 -> bf16 row-major copy of x (4 elems / thread, b64 stores) ----
__global__ void cvt_bf16(const float* __restrict__ src, __bf16* __restrict__ dst,
                         size_t n4) {
  const size_t t = (size_t)blockIdx.x * blockDim.x + threadIdx.x;
  if (t < n4) {
    const float4 v = ((const float4*)src)[t];
    v4bf o; o[0] = f2bf(v.x); o[1] = f2bf(v.y); o[2] = f2bf(v.z); o[3] = f2bf(v.w);
    ((v4bf*)dst)[t] = o;
  }
}

// -------- pack weight [256 x nTiles*16] f32 into per-lane WMMA B fragments --
// layout: out[(((kb/32)*nTiles + ntile)*32 + lane)*16 + e], contiguous per lane
__global__ void pack_w(const float* __restrict__ W, int nTiles,
                       __bf16* __restrict__ out) {
  const int t = blockIdx.x * blockDim.x + threadIdx.x;   // 8*nTiles*32 threads
  if (t >= 8 * nTiles * 32) return;
  const int lane  = t & 31;
  const int ntile = (t >> 5) % nTiles;
  const int kbI   = (t >> 5) / nTiles;
  const int half  = lane >> 4, l15 = lane & 15;
  const int nCols = nTiles * 16;
  const int n     = ntile * 16 + l15;
  __bf16* o = out + (size_t)t * 16;
#pragma unroll
  for (int e = 0; e < 16; ++e) {
    const int k = kbI * 32 + half * 8 + ((e < 8) ? e : (e + 8));
    o[e] = f2bf(W[(size_t)k * nCols + n]);
  }
}

// -------- GEMM1: hb = bf16( x @ W1 + b1 )  (2 M-tiles per wave, B reused) ---
// grid (3125, 2), block 256 (8 waves): wave w -> n-tile blockIdx.y*8+w
__global__ void gemm1_wmma(const __bf16* __restrict__ xb,
                           const __bf16* __restrict__ pW1,
                           const float* __restrict__ b1,
                           __bf16* __restrict__ hb) {
  const int lane  = threadIdx.x & 31;
  const int wave  = threadIdx.x >> 5;
  const int ntile = blockIdx.y * 8 + wave;               // 0..15
  const int m0    = blockIdx.x * 32;                     // two 16-row tiles
  const int half  = lane >> 4, l15 = lane & 15;
  const size_t rowA0 = (size_t)(m0 + l15) * D_IN;
  const size_t rowA1 = (size_t)(m0 + 16 + l15) * D_IN;

  v8f c0 = {}, c1 = {};
#pragma unroll
  for (int kb = 0; kb < D_IN; kb += 32) {
    const int kOff = kb + half * 8;
    const v8bf a0l = *(const v8bf*)(xb + rowA0 + kOff);
    const v8bf a0h = *(const v8bf*)(xb + rowA0 + kOff + 16);
    const v8bf a1l = *(const v8bf*)(xb + rowA1 + kOff);
    const v8bf a1h = *(const v8bf*)(xb + rowA1 + kOff + 16);
    const v16bf b  = *(const v16bf*)(pW1 +
        ((((size_t)(kb >> 5) * 16 + ntile) * 32 + lane) << 4));
    const v16bf a0 = CAT16(a0l, a0h);
    const v16bf a1 = CAT16(a1l, a1h);
    c0 = WMMA_BF16(a0, b, c0);
    c1 = WMMA_BF16(a1, b, c1);
  }
  const int col  = ntile * 16 + l15;
  const float bias = b1[col];
#pragma unroll
  for (int r = 0; r < 8; ++r) {
    const int m = m0 + r + half * 8;                     // C/D: M = r + 8*(lane>=16)
    hb[(size_t)m * HID + col]        = f2bf(c0[r] + bias);
    hb[(size_t)(m + 16) * HID + col] = f2bf(c1[r] + bias);
  }
}

// -------- BatchNorm stats (from bf16 h, f32 accumulation) -------------------
__global__ void bn_zero(float* __restrict__ sum, float* __restrict__ sq) {
  sum[threadIdx.x] = 0.0f; sq[threadIdx.x] = 0.0f;
}
__global__ void bn_stats(const __bf16* __restrict__ hb,
                         float* __restrict__ sum, float* __restrict__ sq) {
  const int c = threadIdx.x;                  // column per thread: rows coalesced
  float s = 0.0f, q = 0.0f;
  for (int r = blockIdx.x; r < N_NODES; r += gridDim.x) {
    const float v = bf2f(hb[(size_t)r * HID + c]);
    s += v; q += v * v;
  }
  atomicAdd(&sum[c], s);
  atomicAdd(&sq[c], q);
}
__global__ void bn_finalize(const float* __restrict__ sum, const float* __restrict__ sq,
                            const float* __restrict__ gamma, const float* __restrict__ beta,
                            float* __restrict__ sOut, float* __restrict__ tOut) {
  const int c = threadIdx.x;
  const float mean = sum[c] * (1.0f / (float)N_NODES);
  const float var  = sq[c] * (1.0f / (float)N_NODES) - mean * mean;
  const float s    = gamma[c] * rsqrtf(var + BN_EPS);
  sOut[c] = s;
  tOut[c] = beta[c] - mean * s;
}
__global__ void w2_scale(const float* __restrict__ W2, const float* __restrict__ s,
                         float* __restrict__ W2f) {
  const int i = blockIdx.x * blockDim.x + threadIdx.x;
  if (i < HID * D_OUT) W2f[i] = s[i / D_OUT] * W2[i];
}
__global__ void b2_fold(const float* __restrict__ W2, const float* __restrict__ t,
                        const float* __restrict__ b2, float* __restrict__ b2f) {
  const int n = threadIdx.x;                  // 64 threads
  float acc = b2[n];
  for (int k = 0; k < HID; ++k) acc += t[k] * W2[(size_t)k * D_OUT + n];
  b2f[n] = acc;
}

// -------- GEMM2: z0 = h @ W2f + b2f (BN folded; 2 M-tiles per wave) ---------
// grid (3125), block 128 (4 waves): wave w -> n-tile w (64 cols total)
__global__ void gemm2_wmma(const __bf16* __restrict__ hb,
                           const __bf16* __restrict__ pW2,
                           const float* __restrict__ b2f,
                           float* __restrict__ z) {
  const int lane  = threadIdx.x & 31;
  const int ntile = threadIdx.x >> 5;                    // 0..3
  const int m0    = blockIdx.x * 32;
  const int half  = lane >> 4, l15 = lane & 15;
  const size_t rowA0 = (size_t)(m0 + l15) * HID;
  const size_t rowA1 = (size_t)(m0 + 16 + l15) * HID;

  v8f c0 = {}, c1 = {};
#pragma unroll
  for (int kb = 0; kb < HID; kb += 32) {
    const int kOff = kb + half * 8;
    const v8bf a0l = *(const v8bf*)(hb + rowA0 + kOff);
    const v8bf a0h = *(const v8bf*)(hb + rowA0 + kOff + 16);
    const v8bf a1l = *(const v8bf*)(hb + rowA1 + kOff);
    const v8bf a1h = *(const v8bf*)(hb + rowA1 + kOff + 16);
    const v16bf b  = *(const v16bf*)(pW2 +
        ((((size_t)(kb >> 5) * 4 + ntile) * 32 + lane) << 4));
    const v16bf a0 = CAT16(a0l, a0h);
    const v16bf a1 = CAT16(a1l, a1h);
    c0 = WMMA_BF16(a0, b, c0);
    c1 = WMMA_BF16(a1, b, c1);
  }
  const int col  = ntile * 16 + l15;
  const float bias = b2f[col];
#pragma unroll
  for (int r = 0; r < 8; ++r) {
    const int m = m0 + r + half * 8;
    z[(size_t)m * D_OUT + col]        = c0[r] + bias;
    z[(size_t)(m + 16) * D_OUT + col] = c1[r] + bias;
  }
}

// -------- GCN normalization (degrees incl. self-loop) -----------------------
__global__ void deg_init(float* __restrict__ deg) {
  const int i = blockIdx.x * blockDim.x + threadIdx.x;
  if (i < N_NODES) deg[i] = 1.0f;             // self-loop contribution
}
__global__ void deg_accum(const int* __restrict__ col, int E, float* __restrict__ deg) {
  const int e = blockIdx.x * blockDim.x + threadIdx.x;
  if (e < E) atomicAdd(&deg[col[e]], 1.0f);
}
__global__ void deg_inv(float* __restrict__ deg) {
  const int i = blockIdx.x * blockDim.x + threadIdx.x;
  if (i < N_NODES) deg[i] = rsqrtf(deg[i]);
}

// -------- APPNP hop: teleport + self-loop, then edge scatter ----------------
__global__ void prop_init(const float* __restrict__ z0, const float* __restrict__ carry,
                          const float* __restrict__ dinv, float* __restrict__ nxt) {
  const size_t t = (size_t)blockIdx.x * blockDim.x + threadIdx.x;
  if (t < (size_t)N_NODES * D_OUT) {
    const int i = (int)(t >> 6);
    const float di = dinv[i];
    nxt[t] = ALPHA * z0[t] + (1.0f - ALPHA) * di * di * carry[t];
  }
}
__global__ void prop_edges(const int* __restrict__ row, const int* __restrict__ col,
                           int E, const float* __restrict__ carry,
                           const float* __restrict__ dinv, float* __restrict__ nxt) {
  const size_t t = (size_t)blockIdx.x * blockDim.x + threadIdx.x;
  if (t < (size_t)E * D_OUT) {
    const int e = (int)(t >> 6);
    const int f = (int)(t & 63);              // 64 lanes share an edge -> coalesced
    if (f == 0 && e + 1024 < E) {             // pull index stream ahead of the atomics
      __builtin_prefetch(&row[e + 1024], 0, 1);
      __builtin_prefetch(&col[e + 1024], 0, 1);
    }
    const int r = row[e], cc = col[e];
    const float w = (1.0f - ALPHA) * dinv[r] * dinv[cc];
    atomicAdd(&nxt[(size_t)cc * D_OUT + f], w * carry[(size_t)r * D_OUT + f]);
  }
}

// -------- log_softmax (one wave32 per row) + emit zk ------------------------
__global__ void logsoftmax_out(const float* __restrict__ zk, float* __restrict__ out) {
  const int rowI = blockIdx.x * (blockDim.x >> 5) + (threadIdx.x >> 5);
  const int lane = threadIdx.x & 31;
  if (rowI >= N_NODES) return;
  const size_t base = (size_t)rowI * D_OUT;
  const float v0 = zk[base + lane];
  const float v1 = zk[base + lane + 32];
  float m = fmaxf(v0, v1);
  for (int off = 16; off > 0; off >>= 1) m = fmaxf(m, __shfl_xor(m, off, 32));
  float s = expf(v0 - m) + expf(v1 - m);
  for (int off = 16; off > 0; off >>= 1) s += __shfl_xor(s, off, 32);
  const float lse = m + logf(s);
  out[base + lane]      = v0 - lse;
  out[base + lane + 32] = v1 - lse;
  const size_t emb = (size_t)N_NODES * D_OUT + base;     // second output: zk
  out[emb + lane]      = v0;
  out[emb + lane + 32] = v1;
}

extern "C" void kernel_launch(void* const* d_in, const int* in_sizes, int n_in,
                              void* d_out, int out_size, void* d_ws, size_t ws_size,
                              hipStream_t stream) {
  const float* x     = (const float*)d_in[0];
  const int*   eidx  = (const int*)d_in[1];
  const float* W1    = (const float*)d_in[2];
  const float* b1    = (const float*)d_in[3];
  const float* gamma = (const float*)d_in[4];
  const float* beta  = (const float*)d_in[5];
  const float* W2    = (const float*)d_in[6];
  const float* b2    = (const float*)d_in[7];
  float* out = (float*)d_out;
  const int E = in_sizes[1] / 2;
  const int* row = eidx;                      // sources
  const int* col = eidx + E;                  // targets

  // -------- workspace carve-out (~180 MB), all regions written before read --
  float* ws = (float*)d_ws;
  size_t off = 0;                             // in floats; all chunks 64-aligned
  float* z0   = ws + off; off += (size_t)N_NODES * D_OUT;
  float* za   = ws + off; off += (size_t)N_NODES * D_OUT;
  float* zb   = ws + off; off += (size_t)N_NODES * D_OUT;
  float* dinv = ws + off; off += (N_NODES + 63) & ~63ull;
  float* sum  = ws + off; off += HID;
  float* sq   = ws + off; off += HID;
  float* sS   = ws + off; off += HID;
  float* tT   = ws + off; off += HID;
  float* W2f  = ws + off; off += (size_t)HID * D_OUT;
  float* b2f  = ws + off; off += 64;
  __bf16* bfbase = (__bf16*)(ws + off);
  size_t boff = 0;                            // in bf16 elems
  __bf16* xb  = bfbase + boff; boff += (size_t)N_NODES * D_IN;
  __bf16* hb  = bfbase + boff; boff += (size_t)N_NODES * HID;
  __bf16* pW1 = bfbase + boff; boff += 8 * 16 * 32 * 16;
  __bf16* pW2 = bfbase + boff; boff += 8 * 4 * 32 * 16;

  const size_t x4 = (size_t)N_NODES * D_IN / 4;
  cvt_bf16<<<(unsigned)((x4 + 255) / 256), 256, 0, stream>>>(x, xb, x4);
  pack_w<<<(8 * 16 * 32 + 255) / 256, 256, 0, stream>>>(W1, 16, pW1);

  gemm1_wmma<<<dim3(N_NODES / 32, 2), 256, 0, stream>>>(xb, pW1, b1, hb);

  bn_zero<<<1, HID, 0, stream>>>(sum, sq);
  bn_stats<<<256, HID, 0, stream>>>(hb, sum, sq);
  bn_finalize<<<1, HID, 0, stream>>>(sum, sq, gamma, beta, sS, tT);
  w2_scale<<<(HID * D_OUT + 255) / 256, 256, 0, stream>>>(W2, sS, W2f);
  b2_fold<<<1, D_OUT, 0, stream>>>(W2, tT, b2, b2f);
  pack_w<<<(8 * 4 * 32 + 255) / 256, 256, 0, stream>>>(W2f, 4, pW2);

  gemm2_wmma<<<N_NODES / 32, 128, 0, stream>>>(hb, pW2, b2f, z0);

  deg_init<<<(N_NODES + 255) / 256, 256, 0, stream>>>(dinv);
  deg_accum<<<(E + 255) / 256, 256, 0, stream>>>(col, E, dinv);
  deg_inv<<<(N_NODES + 255) / 256, 256, 0, stream>>>(dinv);

  const size_t nodeElems = (size_t)N_NODES * D_OUT;
  const size_t edgeElems = (size_t)E * D_OUT;
  const float* carry = z0;
  float* bufs[2] = {za, zb};
  for (int k = 0; k < K_HOPS; ++k) {
    float* nxt = bufs[k & 1];
    prop_init<<<(unsigned)((nodeElems + 255) / 256), 256, 0, stream>>>(z0, carry, dinv, nxt);
    prop_edges<<<(unsigned)((edgeElems + 255) / 256), 256, 0, stream>>>(row, col, E, carry, dinv, nxt);
    carry = nxt;
  }

  logsoftmax_out<<<(N_NODES + 7) / 8, 256, 0, stream>>>(carry, out);
}
